// VanillaMultiHeadAttentionModule_31817117729150
// MI455X (gfx1250) — compile-verified, matrix operand order checked by reading, not run on hardware
//
#include <hip/hip_runtime.h>
#include <cstdint>

typedef __attribute__((ext_vector_type(16))) _Float16 v16h;
typedef __attribute__((ext_vector_type(8)))  float    v8f;
typedef __attribute__((ext_vector_type(4)))  unsigned int v4u;
typedef __attribute__((ext_vector_type(4)))  int v4i;

union HFrag { v16h h; v4u u[2]; };

#define D_MODEL 1024
#define NH      16
#define DH      64
#define SEQ     2048
#define BATCH   4
#define MROWS   (BATCH*SEQ)   /* 8192 */
#define BHEADS  (BATCH*NH)    /* 64   */

#define AS1 __attribute__((address_space(1)))
#define AS3 __attribute__((address_space(3)))

#if defined(__AMDGCN__) && !__has_builtin(__builtin_amdgcn_global_load_async_to_lds_b128)
#warning "gfx1250 async-to-LDS builtin NOT available; using sync VGPR staging fallback"
#endif

// 16-byte global -> LDS copy. Prefers the gfx1250 async DMA path (ASYNCcnt).
__device__ __forceinline__ void async_copy16(const _Float16* g, _Float16* l)
{
#if __has_builtin(__builtin_amdgcn_global_load_async_to_lds_b128)
    __builtin_amdgcn_global_load_async_to_lds_b128(
        (AS1 v4i*)(uintptr_t)g,
        (AS3 v4i*)(unsigned)(uintptr_t)l,
        0, 0);
#else
    *(v4u*)l = *(const v4u*)g;
#endif
}

__device__ __forceinline__ void async_wait_all()
{
#if __has_builtin(__builtin_amdgcn_global_load_async_to_lds_b128)
#if __has_builtin(__builtin_amdgcn_s_wait_asynccnt)
    __builtin_amdgcn_s_wait_asynccnt(0);
#else
    asm volatile("s_wait_asynccnt 0x0" ::: "memory");
#endif
#endif
}

// ---------------------------------------------------------------------------
// LayerNorm (f32 in) -> f16 out. grid (8192, 2): y=0 state->xs, y=1 ctx->xc
// ---------------------------------------------------------------------------
__global__ __launch_bounds__(256)
void ln_kernel(const float* __restrict__ state, const float* __restrict__ context,
               const float* __restrict__ gamma, const float* __restrict__ beta,
               _Float16* __restrict__ xs, _Float16* __restrict__ xc)
{
    const float* src = blockIdx.y ? context : state;
    _Float16*    dst = blockIdx.y ? xc : xs;
    const int row = blockIdx.x;
    const int tid = threadIdx.x;

    const float4 x = *(const float4*)(src + (size_t)row * D_MODEL + tid * 4);
    float s  = x.x + x.y + x.z + x.w;
    float ss = x.x*x.x + x.y*x.y + x.z*x.z + x.w*x.w;
    #pragma unroll
    for (int off = 1; off < 32; off <<= 1) {
        s  += __shfl_xor(s,  off, 32);
        ss += __shfl_xor(ss, off, 32);
    }
    __shared__ float rs_[8], rss_[8];
    const int wid = tid >> 5, lane = tid & 31;
    if (lane == 0) { rs_[wid] = s; rss_[wid] = ss; }
    __syncthreads();
    if (tid == 0) {
        float a = 0.f, b = 0.f;
        #pragma unroll
        for (int i = 0; i < 8; i++) { a += rs_[i]; b += rss_[i]; }
        rs_[0] = a; rss_[0] = b;
    }
    __syncthreads();
    const float mu   = rs_[0] * (1.0f / D_MODEL);
    const float var  = rss_[0] * (1.0f / D_MODEL) - mu * mu;
    const float rinv = rsqrtf(var + 1e-5f);

    const float4 g  = *(const float4*)(gamma + tid * 4);
    const float4 bb = *(const float4*)(beta  + tid * 4);
    _Float16* o = dst + (size_t)row * D_MODEL + tid * 4;
    o[0] = (_Float16)(g.x * (x.x - mu) * rinv + bb.x);
    o[1] = (_Float16)(g.y * (x.y - mu) * rinv + bb.y);
    o[2] = (_Float16)(g.z * (x.z - mu) * rinv + bb.z);
    o[3] = (_Float16)(g.w * (x.w - mu) * rinv + bb.w);
}

// ---------------------------------------------------------------------------
// Weight [K,N] f32 -> [N,K] f16 (transpose + convert). grid(32,32), block(32,8)
// ---------------------------------------------------------------------------
__global__ __launch_bounds__(256)
void wt_kernel(const float* __restrict__ W, _Float16* __restrict__ Wt)
{
    __shared__ float tile[32][33];
    const int kb = blockIdx.x * 32, nb = blockIdx.y * 32;
    const int tx = threadIdx.x, ty = threadIdx.y;
    #pragma unroll
    for (int i = 0; i < 4; i++)
        tile[ty + i * 8][tx] = W[(size_t)(kb + ty + i * 8) * D_MODEL + nb + tx];
    __syncthreads();
    #pragma unroll
    for (int i = 0; i < 4; i++)
        Wt[(size_t)(nb + ty + i * 8) * D_MODEL + kb + tx] = (_Float16)tile[tx][ty + i * 8];
}

// ---------------------------------------------------------------------------
// GEMM: C[8192,1024] = A[8192,1024] (f16, row-major) * Bt[1024,1024]^T (f16, [N][K])
// Double-buffered LDS tiles filled via async global->LDS DMA.
// mode 0: out f16 in [bh][s][64] layout, value = (acc+bias)*scale   (Q/K)
// mode 1: out f16 in [bh][64][s] layout (V transposed)
// mode 2: out f32 row-major, value = acc + bias + resid             (O-proj)
// ---------------------------------------------------------------------------
__global__ __launch_bounds__(256)
void gemm_kernel(const _Float16* __restrict__ A, const _Float16* __restrict__ Bt,
                 const float* __restrict__ bias, void* __restrict__ outp,
                 const float* __restrict__ resid, int mode, float scale)
{
    __shared__ _Float16 As[2][128][32];
    __shared__ _Float16 Bs[2][128][32];

    const int tid  = threadIdx.x;
    const int lane = tid & 31, wid = tid >> 5;
    const int lr   = lane & 15, lh  = lane >> 4;
    const int waveM = wid >> 2, waveN = wid & 3;     // 2 x 4 wave grid
    const int rowBase = blockIdx.y * 128;
    const int colBase = blockIdx.x * 128;

    auto load_tiles = [&](int k0, int bufi) {
        #pragma unroll
        for (int i = 0; i < 2; i++) {
            const int idx = tid + i * 256;
            const int r = idx >> 2, c = (idx & 3) * 8;
            async_copy16(A  + (size_t)(rowBase + r) * D_MODEL + k0 + c, &As[bufi][r][c]);
            async_copy16(Bt + (size_t)(colBase + r) * D_MODEL + k0 + c, &Bs[bufi][r][c]);
        }
    };

    const v8f vzero = {0.f,0.f,0.f,0.f,0.f,0.f,0.f,0.f};
    v8f acc[4][2];
    #pragma unroll
    for (int mi = 0; mi < 4; mi++)
        #pragma unroll
        for (int nj = 0; nj < 2; nj++) acc[mi][nj] = vzero;

    load_tiles(0, 0);
    async_wait_all();
    __syncthreads();

    int buf = 0;
    for (int k0 = 0; k0 < D_MODEL; k0 += 32) {
        if (k0 + 32 < D_MODEL) load_tiles(k0 + 32, buf ^ 1);

        HFrag a[4], b[2];
        #pragma unroll
        for (int mi = 0; mi < 4; mi++) {
            const int r = waveM * 64 + mi * 16 + lr;
            a[mi].u[0] = *(const v4u*)&As[buf][r][lh * 8];
            a[mi].u[1] = *(const v4u*)&As[buf][r][lh * 8 + 16];
        }
        #pragma unroll
        for (int nj = 0; nj < 2; nj++) {
            const int r = waveN * 32 + nj * 16 + lr;
            b[nj].u[0] = *(const v4u*)&Bs[buf][r][lh * 8];
            b[nj].u[1] = *(const v4u*)&Bs[buf][r][lh * 8 + 16];
        }
        #pragma unroll
        for (int mi = 0; mi < 4; mi++)
            #pragma unroll
            for (int nj = 0; nj < 2; nj++)
                acc[mi][nj] = __builtin_amdgcn_wmma_f32_16x16x32_f16(
                    false, a[mi].h, false, b[nj].h, (short)0, acc[mi][nj], false, false);

        async_wait_all();
        __syncthreads();
        buf ^= 1;
    }

    #pragma unroll
    for (int mi = 0; mi < 4; mi++)
    #pragma unroll
    for (int nj = 0; nj < 2; nj++)
    #pragma unroll
    for (int r = 0; r < 8; r++) {
        const int gRow = rowBase + waveM * 64 + mi * 16 + r + 8 * lh;
        const int gCol = colBase + waveN * 32 + nj * 16 + lr;
        const float v = acc[mi][nj][r] + bias[gCol];
        if (mode == 2) {
            float* outF = (float*)outp;
            outF[(size_t)gRow * D_MODEL + gCol] = v + resid[(size_t)gRow * D_MODEL + gCol];
        } else {
            _Float16* outH = (_Float16*)outp;
            const int b = gRow >> 11, s = gRow & 2047;
            const int h = gCol >> 6,  d = gCol & 63;
            if (mode == 0)
                outH[((size_t)(b * NH + h) * SEQ + s) * DH + d] = (_Float16)(v * scale);
            else
                outH[((size_t)(b * NH + h) * DH + d) * SEQ + s] = (_Float16)v;
        }
    }
}

// ---------------------------------------------------------------------------
// Flash attention. Q,K: [bh][s][64] f16 (Q pre-scaled), Vt: [bh][64][s] f16.
// K/V chunks double-buffered in LDS via async DMA, shared by all 8 waves.
// O: [b*S+s][h*64+d] f16. grid(16,64) block 256 (8 waves x 16 q-rows).
// ---------------------------------------------------------------------------
__global__ __launch_bounds__(256)
void flash_kernel(const _Float16* __restrict__ Q, const _Float16* __restrict__ K,
                  const _Float16* __restrict__ Vt, _Float16* __restrict__ O)
{
    __shared__ _Float16 Kbuf[2][64][64];   // 16 KB
    __shared__ _Float16 Vbuf[2][64][64];   // 16 KB
    __shared__ _Float16 Pbuf[8][16][64];   // 16 KB wave-private strips

    const int tid  = threadIdx.x;
    const int lane = tid & 31, wid = tid >> 5;
    const int lr   = lane & 15, lh  = lane >> 4;
    const int bh = blockIdx.y;
    const int s0 = blockIdx.x * 128 + wid * 16;

    const _Float16* Kg0 = K  + (size_t)bh * SEQ * DH;
    const _Float16* Vg0 = Vt + (size_t)bh * DH * SEQ;

    auto load_chunk = [&](int kc, int bufi) {
        #pragma unroll
        for (int i = 0; i < 2; i++) {
            const int idx = tid + i * 256;         // 0..511
            const int r = idx >> 3, seg = (idx & 7) * 8;
            async_copy16(Kg0 + (size_t)(kc + r) * DH + seg,      &Kbuf[bufi][r][seg]);
            async_copy16(Vg0 + (size_t)r * SEQ + kc + seg,       &Vbuf[bufi][r][seg]);
        }
    };

    // Q strip fragments, held in registers for entire key loop
    const _Float16* Qg = Q + ((size_t)bh * SEQ + s0) * DH;
    HFrag qf[2];
    #pragma unroll
    for (int ks = 0; ks < 2; ks++) {
        qf[ks].u[0] = *(const v4u*)(Qg + lr * DH + ks * 32 + lh * 8);
        qf[ks].u[1] = *(const v4u*)(Qg + lr * DH + ks * 32 + lh * 8 + 16);
    }

    const v8f vzero = {0.f,0.f,0.f,0.f,0.f,0.f,0.f,0.f};
    float mrow[8], lrow[8];
    v8f oa[4];
    #pragma unroll
    for (int r = 0; r < 8; r++) { mrow[r] = -3.0e38f; lrow[r] = 0.0f; }
    #pragma unroll
    for (int j = 0; j < 4; j++) oa[j] = vzero;

    _Float16* Pw = &Pbuf[wid][0][0];

    load_chunk(0, 0);
    async_wait_all();
    __syncthreads();

    int buf = 0;
    for (int kc = 0; kc < SEQ; kc += 64) {
        if (kc + 64 < SEQ) load_chunk(kc + 64, buf ^ 1);
        // L2 prefetch two chunks ahead (global_prefetch_b8)
        if (kc + 128 < SEQ) {
            __builtin_prefetch(Kg0 + (size_t)(kc + 128 + (tid >> 2)) * DH, 0, 0);
            __builtin_prefetch(Vg0 + (size_t)(tid & 63) * SEQ + kc + 128, 0, 0);
        }

        // ---- scores: S = Q K^T
        v8f st[4];
        #pragma unroll
        for (int j = 0; j < 4; j++) st[j] = vzero;
        #pragma unroll
        for (int j = 0; j < 4; j++) {
            #pragma unroll
            for (int ks = 0; ks < 2; ks++) {
                HFrag kf;
                kf.u[0] = *(const v4u*)&Kbuf[buf][j * 16 + lr][ks * 32 + lh * 8];
                kf.u[1] = *(const v4u*)&Kbuf[buf][j * 16 + lr][ks * 32 + lh * 8 + 16];
                st[j] = __builtin_amdgcn_wmma_f32_16x16x32_f16(
                    false, qf[ks].h, false, kf.h, (short)0, st[j], false, false);
            }
        }
        // ---- online softmax (row r lives in v8f elem r across the 16-lane half)
        #pragma unroll
        for (int r = 0; r < 8; r++) {
            float mx = fmaxf(fmaxf(st[0][r], st[1][r]), fmaxf(st[2][r], st[3][r]));
            #pragma unroll
            for (int off = 1; off < 16; off <<= 1) mx = fmaxf(mx, __shfl_xor(mx, off, 16));
            const float mnew = fmaxf(mrow[r], mx);
            const float sc = __expf(mrow[r] - mnew);
            mrow[r] = mnew;
            float rsum = 0.0f;
            #pragma unroll
            for (int j = 0; j < 4; j++) {
                const float p = __expf(st[j][r] - mnew);
                st[j][r] = p;
                rsum += p;
            }
            #pragma unroll
            for (int off = 1; off < 16; off <<= 1) rsum += __shfl_xor(rsum, off, 16);
            lrow[r] = lrow[r] * sc + rsum;
            #pragma unroll
            for (int j = 0; j < 4; j++) oa[j][r] *= sc;
        }
        // ---- stage P (C-layout -> A-fragment layout) through wave-private LDS
        #pragma unroll
        for (int j = 0; j < 4; j++)
            #pragma unroll
            for (int r = 0; r < 8; r++)
                Pw[(r + 8 * lh) * 64 + j * 16 + lr] = (_Float16)st[j][r];
        // ---- O += P V
        #pragma unroll
        for (int ks = 0; ks < 2; ks++) {
            HFrag pa;
            pa.u[0] = *(const v4u*)&Pw[lr * 64 + ks * 32 + lh * 8];
            pa.u[1] = *(const v4u*)&Pw[lr * 64 + ks * 32 + lh * 8 + 16];
            #pragma unroll
            for (int j = 0; j < 4; j++) {
                HFrag vf;
                vf.u[0] = *(const v4u*)&Vbuf[buf][j * 16 + lr][ks * 32 + lh * 8];
                vf.u[1] = *(const v4u*)&Vbuf[buf][j * 16 + lr][ks * 32 + lh * 8 + 16];
                oa[j] = __builtin_amdgcn_wmma_f32_16x16x32_f16(
                    false, pa.h, false, vf.h, (short)0, oa[j], false, false);
            }
        }

        async_wait_all();
        __syncthreads();
        buf ^= 1;
    }

    const int b = bh >> 4, h = bh & 15;
    float inv[8];
    #pragma unroll
    for (int r = 0; r < 8; r++) inv[r] = 1.0f / lrow[r];
    #pragma unroll
    for (int j = 0; j < 4; j++)
        #pragma unroll
        for (int r = 0; r < 8; r++) {
            const int s   = s0 + r + 8 * lh;
            const int col = h * DH + j * 16 + lr;
            O[(size_t)(b * SEQ + s) * D_MODEL + col] = (_Float16)(oa[j][r] * inv[r]);
        }
}

// ---------------------------------------------------------------------------
extern "C" void kernel_launch(void* const* d_in, const int* in_sizes, int n_in,
                              void* d_out, int out_size, void* d_ws, size_t ws_size,
                              hipStream_t stream)
{
    (void)in_sizes; (void)n_in; (void)out_size; (void)ws_size;
    const float* state   = (const float*)d_in[0];
    const float* context = (const float*)d_in[1];
    const float* Wq = (const float*)d_in[2];  const float* bq = (const float*)d_in[3];
    const float* Wk = (const float*)d_in[4];  const float* bk = (const float*)d_in[5];
    const float* Wv = (const float*)d_in[6];  const float* bv = (const float*)d_in[7];
    const float* Wo = (const float*)d_in[8];  const float* bo = (const float*)d_in[9];
    const float* gamma = (const float*)d_in[10];
    const float* beta  = (const float*)d_in[11];

    char* ws = (char*)d_ws;
    _Float16* xs  = (_Float16*)(ws);                          // 16 MB
    _Float16* xc  = (_Float16*)(ws + (size_t)(16u << 20));    // 16 MB
    _Float16* Wqt = (_Float16*)(ws + (size_t)(32u << 20));    //  2 MB
    _Float16* Wkt = (_Float16*)(ws + (size_t)(34u << 20));    //  2 MB
    _Float16* Wvt = (_Float16*)(ws + (size_t)(36u << 20));    //  2 MB
    _Float16* Wot = (_Float16*)(ws + (size_t)(38u << 20));    //  2 MB
    _Float16* Qh  = (_Float16*)(ws + (size_t)(40u << 20));    // 16 MB
    _Float16* Kh  = (_Float16*)(ws + (size_t)(56u << 20));    // 16 MB
    _Float16* Vth = (_Float16*)(ws + (size_t)(72u << 20));    // 16 MB
    _Float16* Oh  = (_Float16*)(ws + (size_t)(88u << 20));    // 16 MB (total 104 MB)

    float* out = (float*)d_out;

    ln_kernel<<<dim3(MROWS, 2), 256, 0, stream>>>(state, context, gamma, beta, xs, xc);

    dim3 tb(32, 8);
    wt_kernel<<<dim3(32, 32), tb, 0, stream>>>(Wq, Wqt);
    wt_kernel<<<dim3(32, 32), tb, 0, stream>>>(Wk, Wkt);
    wt_kernel<<<dim3(32, 32), tb, 0, stream>>>(Wv, Wvt);
    wt_kernel<<<dim3(32, 32), tb, 0, stream>>>(Wo, Wot);

    // Q gets the 1/sqrt(64) attention scale folded in
    gemm_kernel<<<dim3(8, 64), 256, 0, stream>>>(xs, Wqt, bq, Qh,  nullptr, 0, 0.125f);
    gemm_kernel<<<dim3(8, 64), 256, 0, stream>>>(xc, Wkt, bk, Kh,  nullptr, 0, 1.0f);
    gemm_kernel<<<dim3(8, 64), 256, 0, stream>>>(xc, Wvt, bv, Vth, nullptr, 1, 1.0f);

    flash_kernel<<<dim3(16, 64), 256, 0, stream>>>(Qh, Kh, Vth, Oh);

    gemm_kernel<<<dim3(8, 64), 256, 0, stream>>>(Oh, Wot, bo, (void*)out, state, 2, 1.0f);
}
